// GAT_62526133895590
// MI455X (gfx1250) — matrix operand order repaired and checked by reading, not compiled, not run on hardware
//
#include <hip/hip_runtime.h>

typedef __attribute__((ext_vector_type(16))) __bf16       v16bf;
typedef __attribute__((ext_vector_type(8)))  float        v8f;
typedef __attribute__((ext_vector_type(8)))  unsigned int v8u;
typedef __attribute__((ext_vector_type(4)))  unsigned int v4u;

#define NNODES 25000
#define NEDGES 400000
#define MT     1563   // ceil(NNODES / 16)

// ---------------- utility kernels ----------------
__global__ void zero_f32(float* p, int n) {
  int i = blockIdx.x * blockDim.x + threadIdx.x;
  if (i < n) p[i] = 0.0f;
}

__global__ void fill_u32(unsigned* p, unsigned v, int n) {
  int i = blockIdx.x * blockDim.x + threadIdx.x;
  if (i < n) p[i] = v;
}

// ---- fused f32->bf16 cast + relayout into WMMA-fragment-major order ----------
// packedA layout: [mtile][kchunk][lane][word0..7], word i of lane l holds the
// bf16 pair (k, k+1) with k = c*32 + (i&4?16:0) + (l>>4)*8 + (i&3)*2, row m = t*16+(l&15).
template <int KC>
__global__ void pack_a(const float* __restrict__ x, unsigned* __restrict__ pA,
                       int M, int total) {
  const int o = blockIdx.x * blockDim.x + threadIdx.x;
  if (o >= total) return;
  constexpr int K = KC * 32;
  const int i = o & 7;
  const int l = (o >> 3) & 31;
  const int c = (o >> 8) % KC;
  const int t = (o >> 8) / KC;
  int m = t * 16 + (l & 15);
  if (m >= M) m = M - 1;                       // clamp; stores guarded in GEMM
  const int k = c * 32 + ((i & 4) ? 16 : 0) + (l >> 4) * 8 + (i & 3) * 2;
  const unsigned lo = __builtin_bit_cast(unsigned short, (__bf16)x[m * K + k]);
  const unsigned hi = __builtin_bit_cast(unsigned short, (__bf16)x[m * K + k + 1]);
  pA[o] = lo | (hi << 16);
}

// packedB layout: [ntile][kchunk][lane][word0..7], word v of lane l holds the
// bf16 pair (k, k+1) with k = c*32 + (v&4?16:0) + (l>>4)*8 + (v&3)*2, column n = u*16+(l&15).
template <int KC>
__global__ void pack_b(const float* __restrict__ W, unsigned* __restrict__ pB,
                       int Nc, int total) {
  const int o = blockIdx.x * blockDim.x + threadIdx.x;
  if (o >= total) return;
  const int v = o & 7;
  const int l = (o >> 3) & 31;
  const int c = (o >> 8) % KC;
  const int u = (o >> 8) / KC;
  const int n = u * 16 + (l & 15);
  const int k = c * 32 + ((v & 4) ? 16 : 0) + (l >> 4) * 8 + (v & 3) * 2;
  const unsigned lo = __builtin_bit_cast(unsigned short, (__bf16)W[k * Nc + n]);
  const unsigned hi = __builtin_bit_cast(unsigned short, (__bf16)W[(k + 1) * Nc + n]);
  pB[o] = lo | (hi << 16);
}

// ---------------- WMMA GEMM ---------------------------------------------------
// 4 waves/block; wave w computes tile (blockIdx.x*16, (blockIdx.y*4+w)*16).
// Each wave async-copies its packed-B tile (KC*1KB) into LDS, waits on
// ASYNCcnt, then runs the fully unrolled K-loop from LDS (B) + global (A).
template <int KC>
__global__ __launch_bounds__(128)
void gemm_bf16_wmma(const unsigned* __restrict__ pA, const unsigned* __restrict__ pB,
                    float* __restrict__ C, int M, int Nc) {
  __shared__ __align__(16) unsigned ldsB[4 * 8 * 32 * 8];   // 32 KB, [wave][c][lane][v]
  const int lane = threadIdx.x & 31;
  const int wave = threadIdx.x >> 5;
  const int t = blockIdx.x;
  const int u = blockIdx.y * 4 + wave;

  // --- async stage of this wave's B tile: 32 B per lane per k-chunk ---
#pragma unroll
  for (int c = 0; c < KC; ++c) {
    const unsigned* gaddr = pB + ((u * KC + c) * 32 + lane) * 8;
    const unsigned ldsByte =
        (unsigned)(unsigned long long)&ldsB[((wave * 8 + c) * 32 + lane) * 8];
    asm volatile("global_load_async_to_lds_b128 %0, %1, off"
                 :: "v"(ldsByte), "v"(gaddr) : "memory");
    asm volatile("global_load_async_to_lds_b128 %0, %1, off offset:16"
                 :: "v"(ldsByte), "v"(gaddr) : "memory");
  }
  asm volatile("s_wait_asynccnt 0" ::: "memory");

  v8f acc = {};
#pragma unroll
  for (int c = 0; c < KC; ++c) {
    const v4u* ga = (const v4u*)(pA + ((t * KC + c) * 32 + lane) * 8);
    const v4u a_lo = ga[0], a_hi = ga[1];
    const v8u au = __builtin_shufflevector(a_lo, a_hi, 0, 1, 2, 3, 4, 5, 6, 7);

    const v4u* lb = (const v4u*)&ldsB[((wave * 8 + c) * 32 + lane) * 8];
    const v4u b_lo = lb[0], b_hi = lb[1];
    const v8u bu = __builtin_shufflevector(b_lo, b_hi, 0, 1, 2, 3, 4, 5, 6, 7);

    acc = __builtin_amdgcn_wmma_f32_16x16x32_bf16(
        false, __builtin_bit_cast(v16bf, au),
        false, __builtin_bit_cast(v16bf, bu),
        (short)0, acc, false, false);
  }

  // C/D layout: word r -> row r (lanes 0-15) / row 8+r (lanes 16-31), col lane&15
  const int m0 = t * 16;
  const int nC = u * 16 + (lane & 15);
  const int rbase = (lane >> 4) * 8;
  float* cp = C + nC;
  if (m0 + 16 <= M) {
#pragma unroll
    for (int r = 0; r < 8; ++r) cp[(m0 + rbase + r) * Nc] = acc[r];
  } else {
#pragma unroll
    for (int r = 0; r < 8; ++r) {
      const int row = m0 + rbase + r;
      if (row < M) cp[row * Nc] = acc[r];
    }
  }
}

// ---------------- el/er = sum(feat * a) per (node, head); d == 64 -------------
template <int H>
__global__ __launch_bounds__(256)
void attn_scores(const float* __restrict__ feat, const float* __restrict__ al,
                 const float* __restrict__ ar, float* __restrict__ el,
                 float* __restrict__ er, int n) {
  const int w    = blockIdx.x * (blockDim.x >> 5) + (threadIdx.x >> 5);
  const int lane = threadIdx.x & 31;
  if (w >= n * H) return;
  const int node = w / H, head = w - node * H;
  const float* f = feat + node * (H * 64) + head * 64;
  const float x0 = f[lane], x1 = f[lane + 32];
  float sl = x0 * al[head * 64 + lane] + x1 * al[head * 64 + lane + 32];
  float sr = x0 * ar[head * 64 + lane] + x1 * ar[head * 64 + lane + 32];
  for (int off = 16; off > 0; off >>= 1) {
    sl += __shfl_xor(sl, off, 32);
    sr += __shfl_xor(sr, off, 32);
  }
  if (lane == 0) { el[w] = sl; er[w] = sr; }
}

// ---------------- edge score + segment max (monotonic uint encoding) ----------
template <int H>
__global__ void edge_scores(const int* __restrict__ src, const int* __restrict__ dst,
                            const float* __restrict__ el, const float* __restrict__ er,
                            float* __restrict__ scores, unsigned* __restrict__ emaxU,
                            int E) {
  const int t = blockIdx.x * blockDim.x + threadIdx.x;
  if (t >= E * H) return;
  const int e = t / H, hd = t - e * H;
  const int s = src[e], d = dst[e];
  float v = el[s * H + hd] + er[d * H + hd];
  v = v > 0.0f ? v : 0.2f * v;                         // leaky_relu
  scores[t] = v;
  const int bits = __float_as_int(v);
  const unsigned enc = (bits >= 0) ? ((unsigned)bits | 0x80000000u) : ~(unsigned)bits;
  atomicMax(&emaxU[d * H + hd], enc);
}

__global__ void emax_finalize(const unsigned* __restrict__ emaxU,
                              float* __restrict__ emaxF, int n) {
  const int i = blockIdx.x * blockDim.x + threadIdx.x;
  if (i >= n) return;
  const unsigned u = emaxU[i];
  const unsigned bits = (u & 0x80000000u) ? (u ^ 0x80000000u) : ~u;
  const float v = __int_as_float((int)bits);
  const bool fin = (v == v) && (fabsf(v) <= 3.0e38f);  // isolated node -> 0
  emaxF[i] = fin ? v : 0.0f;
}

// ---------------- ee = exp(e - emax[dst]); denom = segment_sum ----------------
template <int H>
__global__ void edge_exp(const int* __restrict__ dst, const float* __restrict__ emaxF,
                         float* __restrict__ scores, float* __restrict__ denom, int E) {
  const int t = blockIdx.x * blockDim.x + threadIdx.x;
  if (t >= E * H) return;
  const int e = t / H, hd = t - e * H;
  const int d = dst[e];
  const float ee = __expf(scores[t] - emaxF[d * H + hd]);
  scores[t] = ee;
  atomicAdd(&denom[d * H + hd], ee);
}

// ---------------- agg[dst] += alpha * feat[src]; wave per edge ----------------
template <int H>
__global__ __launch_bounds__(256)
void aggregate(const int* __restrict__ src, const int* __restrict__ dst,
               const float* __restrict__ feat, const float* __restrict__ scores,
               const float* __restrict__ denom, float* __restrict__ agg, int E) {
  const int lane = threadIdx.x & 31;
  const int e = blockIdx.x * (blockDim.x >> 5) + (threadIdx.x >> 5);
  if (e >= E) return;
  const int s = src[e], d = dst[e];
  constexpr int HD = H * 64;
  __builtin_prefetch(feat + s * HD + lane, 0, 1);      // global_prefetch_b8
#pragma unroll
  for (int jj = 0; jj < HD / 32; ++jj) {
    const int j = lane + jj * 32;
    const int head = j >> 6;
    const float w = scores[e * H + head] / fmaxf(denom[d * H + head], 1e-9f);
    atomicAdd(&agg[d * HD + j], w * feat[s * HD + j]);
  }
}

// ---------------- epilogues ---------------------------------------------------
__global__ void elu_k(const float* __restrict__ in, float* __restrict__ out, int n) {
  const int i = blockIdx.x * blockDim.x + threadIdx.x;
  if (i >= n) return;
  const float x = in[i];
  out[i] = x > 0.0f ? x : (__expf(x) - 1.0f);
}

__global__ void elu_res_k(const float* __restrict__ in, const float* __restrict__ res,
                          float* __restrict__ out, int n) {
  const int i = blockIdx.x * blockDim.x + threadIdx.x;
  if (i >= n) return;
  const float x = in[i];
  out[i] = (x > 0.0f ? x : (__expf(x) - 1.0f)) + res[i];
}

__global__ void mean_heads(const float* __restrict__ agg, float* __restrict__ out, int n) {
  const int t = blockIdx.x * blockDim.x + threadIdx.x;
  if (t >= n * 64) return;
  const int node = t >> 6, j = t & 63;
  const float* p = agg + node * 384;
  float s = 0.0f;
#pragma unroll
  for (int head = 0; head < 6; ++head) s += p[head * 64 + j];
  out[t] = s * (1.0f / 6.0f);
}

// ---------------- host orchestration -----------------------------------------
struct Bufs {
  float *feat, *agg, *x0, *el, *er, *emaxF, *denom, *scores;
  unsigned *emaxU, *pA, *pB;
};

static inline int cdiv(int a, int b) { return (a + b - 1) / b; }

template <int H, int KC>
static void run_gat(const float* x, const float* W, const float* al, const float* ar,
                    const int* src, const int* dst, const Bufs& b, hipStream_t stream) {
  constexpr int Nc = H * 64;
  constexpr int NT = Nc / 16;

  const int aTot = MT * KC * 256;
  pack_a<KC><<<cdiv(aTot, 256), 256, 0, stream>>>(x, b.pA, NNODES, aTot);
  const int bTot = NT * KC * 256;
  pack_b<KC><<<cdiv(bTot, 256), 256, 0, stream>>>(W, b.pB, Nc, bTot);

  dim3 g(MT, NT / 4);
  gemm_bf16_wmma<KC><<<g, 128, 0, stream>>>(b.pA, b.pB, b.feat, NNODES, Nc);

  const int nh = NNODES * H;
  attn_scores<H><<<cdiv(nh, 8), 256, 0, stream>>>(b.feat, al, ar, b.el, b.er, NNODES);
  fill_u32<<<cdiv(nh, 256), 256, 0, stream>>>(b.emaxU, 0u, nh);
  zero_f32<<<cdiv(nh, 256), 256, 0, stream>>>(b.denom, nh);

  const int eh = NEDGES * H;
  edge_scores<H><<<cdiv(eh, 256), 256, 0, stream>>>(src, dst, b.el, b.er, b.scores,
                                                    b.emaxU, NEDGES);
  emax_finalize<<<cdiv(nh, 256), 256, 0, stream>>>(b.emaxU, b.emaxF, nh);
  edge_exp<H><<<cdiv(eh, 256), 256, 0, stream>>>(dst, b.emaxF, b.scores, b.denom, NEDGES);

  const int na = NNODES * Nc;
  zero_f32<<<cdiv(na, 256), 256, 0, stream>>>(b.agg, na);
  aggregate<H><<<cdiv(NEDGES, 8), 256, 0, stream>>>(src, dst, b.feat, b.scores,
                                                    b.denom, b.agg, NEDGES);
}

extern "C" void kernel_launch(void* const* d_in, const int* in_sizes, int n_in,
                              void* d_out, int out_size, void* d_ws, size_t ws_size,
                              hipStream_t stream) {
  (void)in_sizes; (void)n_in; (void)out_size; (void)ws_size;
  const float* feats = (const float*)d_in[0];
  const int*   src   = (const int*)d_in[1];
  const int*   dst   = (const int*)d_in[2];
  const float* W0    = (const float*)d_in[3];
  const float* al0   = (const float*)d_in[4];
  const float* ar0   = (const float*)d_in[5];
  const float* W1    = (const float*)d_in[6];
  const float* al1   = (const float*)d_in[7];
  const float* ar1   = (const float*)d_in[8];
  const float* W2    = (const float*)d_in[9];
  const float* al2   = (const float*)d_in[10];
  const float* ar2   = (const float*)d_in[11];
  float* out = (float*)d_out;

  char* ws = (char*)d_ws;
  size_t off = 0;
  auto carve = [&](size_t bytes) -> void* {
    void* p = ws + off;
    off += (bytes + 255) & ~(size_t)255;
    return p;
  };

  Bufs b;
  b.feat   = (float*)   carve((size_t)NNODES * 384 * 4);
  b.agg    = (float*)   carve((size_t)NNODES * 384 * 4);
  b.x0     = (float*)   carve((size_t)NNODES * 256 * 4);
  b.pA     = (unsigned*)carve((size_t)MT * 8 * 256 * 4);
  b.pB     = (unsigned*)carve((size_t)24 * 8 * 256 * 4);
  b.el     = (float*)   carve((size_t)NNODES * 6 * 4);
  b.er     = (float*)   carve((size_t)NNODES * 6 * 4);
  b.emaxF  = (float*)   carve((size_t)NNODES * 6 * 4);
  b.denom  = (float*)   carve((size_t)NNODES * 6 * 4);
  b.emaxU  = (unsigned*)carve((size_t)NNODES * 6 * 4);
  b.scores = (float*)   carve((size_t)NEDGES * 6 * 4);

  // layer 0: elu, flatten heads
  run_gat<4, 4>(feats, W0, al0, ar0, src, dst, b, stream);
  elu_k<<<cdiv(NNODES * 256, 256), 256, 0, stream>>>(b.agg, b.x0, NNODES * 256);

  // layer 1: elu + residual (in-place elementwise is safe)
  run_gat<4, 8>(b.x0, W1, al1, ar1, src, dst, b, stream);
  elu_res_k<<<cdiv(NNODES * 256, 256), 256, 0, stream>>>(b.agg, b.x0, b.x0,
                                                         NNODES * 256);

  // layer 2: no activation, mean over 6 heads
  run_gat<6, 8>(b.x0, W2, al2, ar2, src, dst, b, stream);
  mean_heads<<<cdiv(NNODES * 64, 256), 256, 0, stream>>>(b.agg, out, NNODES);
}
